// BinaryXnorExceptOutliersLinear_1279900254251
// MI455X (gfx1250) — matrix operand order repaired and checked by reading, not compiled, not run on hardware
//
#include <hip/hip_runtime.h>
#include <hip/hip_bf16.h>

typedef __bf16 bf16_t;
typedef __attribute__((ext_vector_type(16))) __bf16 v16bf;
typedef __attribute__((ext_vector_type(8)))  __bf16 v8bf;
typedef __attribute__((ext_vector_type(8)))  float  v8f;

#define IN_FEATS  8192
#define OUT_FEATS 8192
#define BATCH     32
#define NBINS     65536
#define KSPLIT    4
#define KCHUNK    (IN_FEATS / KSPLIT)   // 2048

struct Scalars {
  double       totalAbs;     // sum |w| over all elements
  double       lowTailAbs;   // sum |w| over buckets strictly below bucketLo
  double       highTailAbs;  // sum |w| over buckets strictly above bucketHi
  unsigned int bucketLo;
  unsigned int bucketHi;
  unsigned int remLo;
  unsigned int remHi;
  float        lower;
  float        upper;
  float        bscale;
};

// monotone (ascending) mapping float -> u32 and back
__device__ __forceinline__ unsigned int f2key(float f) {
  unsigned int b = __float_as_uint(f);
  return b ^ ((b & 0x80000000u) ? 0xFFFFFFFFu : 0x80000000u);
}
__device__ __forceinline__ float key2f(unsigned int k) {
  unsigned int b = k ^ ((k & 0x80000000u) ? 0x80000000u : 0xFFFFFFFFu);
  return __uint_as_float(b);
}

// ---------------- pass 0: zero accumulators (graph-replay determinism) ----
__global__ void k_zero(unsigned int* hist3, Scalars* s) {
  int i = blockIdx.x * blockDim.x + threadIdx.x;
  if (i < 3 * NBINS) hist3[i] = 0u;
  if (i == 0) { s->totalAbs = 0.0; s->lowTailAbs = 0.0; s->highTailAbs = 0.0; }
}

// ---------------- x (f32) -> bf16 -----------------------------------------
__global__ void k_xbf16(const float* __restrict__ x, bf16_t* __restrict__ xb, int n) {
  int i = blockIdx.x * blockDim.x + threadIdx.x;
  if (i < n) xb[i] = (bf16_t)x[i];
}

// ---------------- W pass 1 (fused): per-row min/max + global key histogram
//                  + total sum|w| ------------------------------------------
// 1024 blocks x 256 threads; each of the 8 waves owns one full row (wave32
// shfl reduction for min/max); all waves share one 256 KB LDS histogram.
__global__ void k_pass1(const float* __restrict__ W,
                        float* __restrict__ rowScale,
                        float* __restrict__ rowZp,
                        unsigned int* __restrict__ ghist,
                        Scalars* s) {
  extern __shared__ unsigned int lh[];            // NBINS u32 = 256 KB
  __shared__ double dsum[256];
  const int tid  = threadIdx.x;
  const int lane = tid & 31;
  const int wave = tid >> 5;
  for (int i = tid; i < NBINS; i += 256) lh[i] = 0u;
  __syncthreads();

  const int row = blockIdx.x * 8 + wave;
  const float4* wr4 = (const float4*)(W + (size_t)row * IN_FEATS);
  float  mn = 3.4e38f, mx = -3.4e38f;
  double asum = 0.0;
  for (int i = lane; i < IN_FEATS / 4; i += 32) {
    float4 v = wr4[i];
    float f[4] = {v.x, v.y, v.z, v.w};
    #pragma unroll
    for (int j = 0; j < 4; ++j) {
      mn = fminf(mn, f[j]);
      mx = fmaxf(mx, f[j]);
      asum += (double)fabsf(f[j]);
      atomicAdd(&lh[f2key(f[j]) >> 16], 1u);
    }
  }
  // wave32 min/max reduction
  #pragma unroll
  for (int off = 16; off > 0; off >>= 1) {
    mn = fminf(mn, __shfl_xor(mn, off, 32));
    mx = fmaxf(mx, __shfl_xor(mx, off, 32));
  }
  if (lane == 0) {
    float range = mx - mn;
    rowScale[row] = range * (1.0f / 255.0f);
    rowZp[row]    = rintf(mn - 128.0f * range / 255.0f);
  }
  // block reduce sum|w|
  dsum[tid] = asum;
  __syncthreads();
  for (int st = 128; st > 0; st >>= 1) {
    if (tid < st) dsum[tid] += dsum[tid + st];
    __syncthreads();
  }
  if (tid == 0) atomicAdd(&s->totalAbs, dsum[0]);
  // flush histogram
  for (int i = tid; i < NBINS; i += 256)
    if (lh[i]) atomicAdd(&ghist[i], lh[i]);
}

// ---------------- find buckets holding the two ranks ----------------------
__global__ void k_select1(const unsigned int* __restrict__ hist, Scalars* s,
                          unsigned int rankLo, unsigned int rankHi) {
  __shared__ unsigned long long part[256];
  const int tid = threadIdx.x;
  unsigned long long sum = 0;
  for (int j = 0; j < 256; ++j) sum += hist[tid * 256 + j];
  part[tid] = sum;
  __syncthreads();
  if (tid == 0) {
    for (int pass = 0; pass < 2; ++pass) {
      unsigned long long rank = pass ? (unsigned long long)rankHi
                                     : (unsigned long long)rankLo;
      unsigned long long c = 0;
      int chunk = 0;
      while (chunk < 255 && c + part[chunk] <= rank) { c += part[chunk]; ++chunk; }
      int bin = chunk * 256;
      while (bin < NBINS - 1 && c + hist[bin] <= rank) { c += hist[bin]; ++bin; }
      if (pass) { s->bucketHi = (unsigned)bin; s->remHi = (unsigned)(rank - c); }
      else      { s->bucketLo = (unsigned)bin; s->remLo = (unsigned)(rank - c); }
    }
  }
}

// ---------------- W pass 2 (fused): refine histogram inside boundary
//                  buckets + outlier-tail sum|w| ----------------------------
__global__ void k_pass2(const float* __restrict__ W, Scalars* s,
                        unsigned int* __restrict__ histLo,
                        unsigned int* __restrict__ histHi, int n) {
  __shared__ double dlo[256];
  __shared__ double dhi[256];
  const unsigned int bLo = s->bucketLo, bHi = s->bucketHi;
  const int tid = threadIdx.x;
  const int stride = gridDim.x * blockDim.x;
  const int n4 = n >> 2;
  const float4* W4 = (const float4*)W;
  double lowAbs = 0.0, highAbs = 0.0;
  for (int i = blockIdx.x * blockDim.x + tid; i < n4; i += stride) {
    float4 v = W4[i];
    float f[4] = {v.x, v.y, v.z, v.w};
    #pragma unroll
    for (int j = 0; j < 4; ++j) {
      unsigned int k = f2key(f[j]);
      unsigned int top = k >> 16;
      if (top == bLo)     atomicAdd(&histLo[k & 0xFFFFu], 1u);
      else if (top < bLo) lowAbs += (double)fabsf(f[j]);
      if (top == bHi)      atomicAdd(&histHi[k & 0xFFFFu], 1u);
      else if (top > bHi)  highAbs += (double)fabsf(f[j]);
    }
  }
  dlo[tid] = lowAbs; dhi[tid] = highAbs;
  __syncthreads();
  for (int st = 128; st > 0; st >>= 1) {
    if (tid < st) { dlo[tid] += dlo[tid + st]; dhi[tid] += dhi[tid + st]; }
    __syncthreads();
  }
  if (tid == 0) {
    atomicAdd(&s->lowTailAbs,  dlo[0]);
    atomicAdd(&s->highTailAbs, dhi[0]);
  }
}

// ---------------- exact thresholds + analytic binary_scale ----------------
// Refined bins are EXACT 32-bit float keys, so within-bucket |w| sums are
// count(bin) * |value(bin)| -- no third pass over W needed.
__global__ void k_select2(const unsigned int* __restrict__ histLo,
                          const unsigned int* __restrict__ histHi,
                          Scalars* s, unsigned int rankLo, unsigned int rankHi,
                          unsigned int n) {
  if (threadIdx.x != 0 || blockIdx.x != 0) return;

  // lower threshold
  unsigned long long c = 0; unsigned int bin = 0;
  double partLowAbs = 0.0;
  while (bin < NBINS - 1 && c + histLo[bin] <= s->remLo) {
    partLowAbs += (double)histLo[bin] * (double)fabsf(key2f((s->bucketLo << 16) | bin));
    c += histLo[bin]; ++bin;
  }
  float lower = key2f((s->bucketLo << 16) | bin);
  unsigned long long cntBelow = (unsigned long long)(rankLo - s->remLo) + c;  // #(w < lower)
  double absBelow = s->lowTailAbs + partLowAbs;

  // upper threshold
  c = 0; bin = 0;
  while (bin < NBINS - 1 && c + histHi[bin] <= s->remHi) { c += histHi[bin]; ++bin; }
  unsigned int upBin = bin;
  float upper = key2f((s->bucketHi << 16) | upBin);
  unsigned long long cntLE =
      (unsigned long long)(rankHi - s->remHi) + c + histHi[upBin];            // #(w <= upper)
  double partHighAbs = 0.0;
  for (unsigned int b = upBin + 1; b < NBINS; ++b)
    partHighAbs += (double)histHi[b] * (double)fabsf(key2f((s->bucketHi << 16) | b));
  unsigned long long cntAbove = (unsigned long long)n - cntLE;                // #(w > upper)
  double absAbove = s->highTailAbs + partHighAbs;

  unsigned long long keepCnt = (unsigned long long)n - cntBelow - cntAbove;
  double keepAbs = s->totalAbs - absBelow - absAbove;

  s->lower  = lower;
  s->upper  = upper;
  s->bscale = (float)(keepAbs / (double)keepCnt);
}

// ---------------- W pass 3: fused dequant/binarize + WMMA GEMM (split-K) --
// part[split][m][o] = sum_{k in split} Xb[m,k] * w_sim(W[o,k])
// grid = (OUT_FEATS/128, KSPLIT); block = 256 threads = 8 waves.
// Each wave owns one 16-wide N tile and BOTH 16-row M tiles: every weight
// element is loaded + dequantized exactly once and feeds two chained
// v_wmma_f32_16x16x32_bf16 ops (acc0: M=0-15, acc1: M=16-31).
__global__ void __launch_bounds__(256) k_gemm(
    const float*  __restrict__ W,
    const bf16_t* __restrict__ Xb,
    const float*  __restrict__ rowScale,
    const float*  __restrict__ rowZp,
    const Scalars* __restrict__ s,
    float* __restrict__ part)
{
  const int lane  = threadIdx.x & 31;
  const int wave  = threadIdx.x >> 5;          // 0..7 -> N tile
  const int l15   = lane & 15;
  const int hi    = lane >> 4;                 // 0: lanes 0-15, 1: lanes 16-31
  const int obase = blockIdx.x * 128 + wave * 16;
  const int kbase = blockIdx.y * KCHUNK;

  const int o = obase + l15;
  const float lower  = s->lower;
  const float upper  = s->upper;
  const float bscale = s->bscale;
  const float scale  = rowScale[o];
  const float zp     = rowZp[o];
  const float rinv   = 1.0f / scale;

  // A fragments (16x32 bf16): lane hi-half holds 16 consecutive K values {hi*16..hi*16+15}
  const bf16_t* arow0 = Xb + (size_t)(l15)      * IN_FEATS + hi * 16;  // M = 0..15
  const bf16_t* arow1 = Xb + (size_t)(16 + l15) * IN_FEATS + hi * 16;  // M = 16..31
  // B fragment (32x16 bf16): lane holds 16 consecutive K values of column o
  const float*  wrow  = W  + (size_t)o * IN_FEATS + hi * 16;

  v8f acc0 = {0.f, 0.f, 0.f, 0.f, 0.f, 0.f, 0.f, 0.f};
  v8f acc1 = {0.f, 0.f, 0.f, 0.f, 0.f, 0.f, 0.f, 0.f};

  for (int k0 = kbase; k0 < kbase + KCHUNK; k0 += 32) {
    __builtin_prefetch(wrow + k0 + 256, 0, 1);   // global_prefetch_b8 ahead of stream

    v8bf a0lo = *(const v8bf*)(arow0 + k0);
    v8bf a0hi = *(const v8bf*)(arow0 + k0 + 8);
    v8bf a1lo = *(const v8bf*)(arow1 + k0);
    v8bf a1hi = *(const v8bf*)(arow1 + k0 + 8);
    v16bf a0, a1;
    #pragma unroll
    for (int i = 0; i < 8; ++i) {
      a0[i] = a0lo[i]; a0[8 + i] = a0hi[i];
      a1[i] = a1lo[i]; a1[8 + i] = a1hi[i];
    }

    const float4* wp = (const float4*)(wrow + k0);
    float4 f0 = wp[0], f1 = wp[1], f2 = wp[2], f3 = wp[3];
    float wv[16] = {f0.x, f0.y, f0.z, f0.w, f1.x, f1.y, f1.z, f1.w,
                    f2.x, f2.y, f2.z, f2.w, f3.x, f3.y, f3.z, f3.w};

    v16bf b;
    #pragma unroll
    for (int i = 0; i < 16; ++i) {
      float w  = wv[i];
      float q  = rintf((w - zp) * rinv);
      q        = fminf(fmaxf(q, 0.0f), 255.0f);
      float wq = fmaf(q, scale, zp);                         // dequantized 8-bit value
      float bn = (wq > 0.0f) ? bscale : ((wq < 0.0f) ? -bscale : 0.0f);
      float vs = ((w < lower) || (w > upper)) ? wq : bn;     // outlier ? wq : binarized
      b[i] = (bf16_t)vs;
    }

    acc0 = __builtin_amdgcn_wmma_f32_16x16x32_bf16(
        false, a0, false, b, (short)0, acc0, false, false);
    acc1 = __builtin_amdgcn_wmma_f32_16x16x32_bf16(
        false, a1, false, b, (short)0, acc1, false, false);
  }

  float* pbase = part + (size_t)blockIdx.y * BATCH * OUT_FEATS;
  #pragma unroll
  for (int v = 0; v < 8; ++v) {
    int m0 = v + hi * 8;                        // D layout: VGPR v -> M=v (lo) / v+8 (hi)
    pbase[(size_t)m0 * OUT_FEATS + o]        = acc0[v];
    pbase[(size_t)(16 + m0) * OUT_FEATS + o] = acc1[v];
  }
}

// ---------------- split-K reduce + bias -----------------------------------
__global__ void k_reduce(const float* __restrict__ part,
                         const float* __restrict__ bias,
                         float* __restrict__ out, int n) {
  int i = blockIdx.x * blockDim.x + threadIdx.x;
  if (i >= n) return;
  float v = bias[i & (OUT_FEATS - 1)];
  #pragma unroll
  for (int sp = 0; sp < KSPLIT; ++sp)
    v += part[(size_t)sp * BATCH * OUT_FEATS + i];
  out[i] = v;
}

// --------------------------------------------------------------------------
extern "C" void kernel_launch(void* const* d_in, const int* in_sizes, int n_in,
                              void* d_out, int out_size, void* d_ws, size_t ws_size,
                              hipStream_t stream) {
  (void)n_in; (void)out_size; (void)ws_size;
  const float* x    = (const float*)d_in[0];   // (32,1,8192) f32
  const float* W    = (const float*)d_in[1];   // (8192,8192) f32
  const float* bias = (const float*)d_in[2];   // (8192,)     f32
  float* out = (float*)d_out;                  // (32,1,8192) f32

  const int nW = in_sizes[1];                  // 67108864
  const int nX = in_sizes[0];                  // 262144

  // workspace layout
  char* ws = (char*)d_ws;
  float*        partial = (float*)ws;                                   // 4 MB
  size_t        off     = (size_t)KSPLIT * BATCH * OUT_FEATS * 4;
  bf16_t*       xb      = (bf16_t*)(ws + off);                          // 512 KB
  unsigned int* hist1   = (unsigned int*)(ws + off + 524288);           // 256 KB
  unsigned int* histLo  = hist1 + NBINS;                                // 256 KB
  unsigned int* histHi  = histLo + NBINS;                               // 256 KB
  float*        rowSc   = (float*)(ws + off + 524288 + 3 * NBINS * 4);  // 32 KB
  float*        rowZp   = rowSc + OUT_FEATS;                            // 32 KB
  Scalars*      s       = (Scalars*)(rowZp + OUT_FEATS);

  // torch.kthvalue is 1-indexed: ranks (0-based) of lower/upper thresholds
  double dn = (double)nW;
  unsigned int rankLo = (unsigned int)((long long)(dn * 0.025) - 1);
  unsigned int rankHi = (unsigned int)((long long)(dn * 0.975) - 1);

  k_zero   <<<(3 * NBINS + 255) / 256, 256, 0, stream>>>(hist1, s);
  k_xbf16  <<<(nX + 255) / 256, 256, 0, stream>>>(x, xb, nX);
  k_pass1  <<<OUT_FEATS / 8, 256, NBINS * sizeof(unsigned int), stream>>>(
               W, rowSc, rowZp, hist1, s);
  k_select1<<<1, 256, 0, stream>>>(hist1, s, rankLo, rankHi);
  k_pass2  <<<1024, 256, 0, stream>>>(W, s, histLo, histHi, nW);
  k_select2<<<1, 1, 0, stream>>>(histLo, histHi, s, rankLo, rankHi,
                                 (unsigned int)nW);

  dim3 ggrid(OUT_FEATS / 128, KSPLIT);
  k_gemm   <<<ggrid, 256, 0, stream>>>(W, xb, rowSc, rowZp, s, partial);
  k_reduce <<<(nX + 255) / 256, 256, 0, stream>>>(partial, bias, out, nX);
}
